// LocalFeatureLossLayer_45183055954173
// MI455X (gfx1250) — compile-verified, matrix operand order checked by reading, not run on hardware
//
#include <hip/hip_runtime.h>
#include <hip/hip_bf16.h>
#include <math.h>

typedef __attribute__((ext_vector_type(16))) __bf16 v16bf;
typedef __attribute__((ext_vector_type(8)))  __bf16 v8bf;
typedef __attribute__((ext_vector_type(8)))  float  v8f;

// ---------------------------------------------------------------------------
// Fragment loader for 16-bit WMMA A/B operands (16x16x32 bf16).
// A 16x32 (MxK) layout: lanes 0-15 hold row M=lane, K={0..7,16..23};
// lanes 16-31 hold row M=lane-16, K={8..15,24..31}.  B (32x16, KxN) uses the
// symmetric layout with N across lanes.  `base` is row-major with `stride`
// elements per row; `row` selects the M (or N) line; `k0` is the K base.
// ---------------------------------------------------------------------------
__device__ __forceinline__ v16bf load_frag(const __bf16* base, int stride,
                                           int row, int k0, int khalf) {
  union { v16bf v; v8bf h[2]; } u;
  const __bf16* p = base + row * stride + k0 + khalf * 8;
  u.h[0] = *(const v8bf*)(p);
  u.h[1] = *(const v8bf*)(p + 16);
  return u.v;
}

__global__ void zero_kernel(float* out) {
  if (threadIdx.x == 0) out[0] = 0.0f;
}

// Per-(b,channel) mean / rstd over the N spatial positions (NHWC columns).
__global__ void stats_kernel(const float* __restrict__ x,
                             float* __restrict__ mean, float* __restrict__ rstd,
                             int N, int C) {
  const int b  = blockIdx.x / C;
  const int ch = blockIdx.x % C;
  __shared__ float s1[256];
  __shared__ float s2[256];
  float a = 0.f, q = 0.f;
  for (int n = threadIdx.x; n < N; n += 256) {
    float v = x[((size_t)b * N + n) * C + ch];
    a += v; q += v * v;
  }
  s1[threadIdx.x] = a; s2[threadIdx.x] = q;
  __syncthreads();
  for (int off = 128; off; off >>= 1) {
    if (threadIdx.x < off) {
      s1[threadIdx.x] += s1[threadIdx.x + off];
      s2[threadIdx.x] += s2[threadIdx.x + off];
    }
    __syncthreads();
  }
  if (threadIdx.x == 0) {
    float m = s1[0] / (float)N;
    float v = s2[0] / (float)N - m * m;
    mean[b * C + ch] = m;
    rstd[b * C + ch] = 1.0f / sqrtf(fmaxf(v, 0.f) + 1e-5f);
  }
}

// Per-(b,row) inverse L2 norm of the instance-normalized row (wave per row).
__global__ void rownorm_kernel(const float* __restrict__ x,
                               const float* __restrict__ mean,
                               const float* __restrict__ rstd,
                               float* __restrict__ inv,
                               int N, int C, int rows) {
  const int row = blockIdx.x * 8 + (threadIdx.x >> 5);
  if (row >= rows) return;
  const int lane = threadIdx.x & 31;
  const int b = row / N;
  const float* xr = x + (size_t)row * C;
  float s = 0.f;
  for (int c = lane; c < C; c += 32) {
    float v = (xr[c] - mean[b * C + c]) * rstd[b * C + c];
    s += v * v;
  }
  for (int off = 16; off; off >>= 1) s += __shfl_down(s, off, 32);
  if (lane == 0) inv[row] = 1.0f / fmaxf(sqrtf(s), 1e-12f);
}

// ---------------------------------------------------------------------------
// Attention transfer + loss.  One workgroup (4 wave32) per 16-row Q tile.
// Channel dimension split across the 4 waves; logits reduced via ds_add_f32.
// ---------------------------------------------------------------------------
template <int C>
__launch_bounds__(128, 1)
__global__ void attn_kernel(const float* __restrict__ cs,
                            const float* __restrict__ cf,
                            const float* __restrict__ sf,
                            const float* __restrict__ meanC,
                            const float* __restrict__ rstdC,
                            const float* __restrict__ meanS,
                            const float* __restrict__ rstdS,
                            const float* __restrict__ qinv,
                            const float* __restrict__ kinv,
                            float* __restrict__ out,
                            int N, float scale) {
  constexpr int NW = 4;          // waves per workgroup
  constexpr int Cw = C / NW;     // channels owned per wave
  constexpr int KQ = Cw / 32;    // K-chunks (of 32) per wave for logits
  constexpr int NT = Cw / 16;    // 16-wide output channel tiles per wave

  __shared__ __align__(16) __bf16 sQ[16 * C];   // normalized Q, row-major
  __shared__ __align__(16) __bf16 sK[32 * C];   // normalized K block, row-major
  __shared__ __align__(16) __bf16 sVt[C * 32];  // raw V block, transposed [C][32]
  __shared__ float sLog[16 * 32];               // logits reduction tile
  __shared__ float sDen[16];                    // softmax denominators
  __shared__ float sMs[C];                      // staged style mean
  __shared__ float sRs[C];                      // staged style rstd

  const int tid   = threadIdx.x;
  const int lane  = tid & 31;
  const int w     = tid >> 5;
  const int row16 = lane & 15;
  const int khalf = lane >> 4;
  const int cb    = w * Cw;

  const int tilesb = N / 16;
  const int b  = blockIdx.x / tilesb;
  const int n0 = (blockIdx.x % tilesb) * 16;

  // Stage style stats into LDS.
  for (int i = tid; i < C; i += 128) {
    sMs[i] = meanS[b * C + i];
    sRs[i] = rstdS[b * C + i];
  }
  // Build normalized Q tile (nc * qinv) in bf16.
  for (int i = tid; i < 16 * C; i += 128) {
    int r = i / C, c = i - r * C;
    int n = n0 + r;
    float x  = cf[((size_t)b * N + n) * C + c];
    float qn = (x - meanC[b * C + c]) * rstdC[b * C + c] * qinv[(size_t)b * N + n];
    sQ[i] = (__bf16)qn;
  }
  __syncthreads();

  v16bf qf[KQ];
  for (int kc = 0; kc < KQ; ++kc)
    qf[kc] = load_frag(sQ, C, row16, cb + kc * 32, khalf);

  v8f Macc[NT], Wacc[NT];
  for (int t = 0; t < NT; ++t)
    for (int r = 0; r < 8; ++r) { Macc[t][r] = 0.f; Wacc[t][r] = 0.f; }
  float denom = 0.f;  // meaningful in wave 0 lanes 0-15 only

  const int MB = N / 32;
  for (int mb = 0; mb < MB; ++mb) {
    __syncthreads();  // previous iteration's LDS consumers done

    // Cooperative load: normalized K (row-major) and raw V (transposed).
    for (int i = tid; i < 32 * C; i += 128) {
      int m = i / C, c = i - m * C;
      int gm = mb * 32 + m;
      float x = sf[((size_t)b * N + gm) * C + c];
      sK[i]            = (__bf16)((x - sMs[c]) * sRs[c] * kinv[(size_t)b * N + gm]);
      sVt[c * 32 + m]  = (__bf16)x;
    }
    for (int i = tid; i < 16 * 32; i += 128) sLog[i] = 0.f;
    if (mb + 1 < MB && tid < 32)  // prefetch next style block (one line per row)
      __builtin_prefetch(&sf[((size_t)b * N + (mb + 1) * 32 + tid) * C], 0, 0);
    __syncthreads();

    // Partial logits over this wave's channel slice; cross-wave ds_add_f32.
    for (int t2 = 0; t2 < 2; ++t2) {
      v8f acc;
      for (int r = 0; r < 8; ++r) acc[r] = 0.f;
      for (int kc = 0; kc < KQ; ++kc) {
        v16bf bfm = load_frag(sK, C, t2 * 16 + row16, cb + kc * 32, khalf);
        acc = __builtin_amdgcn_wmma_f32_16x16x32_bf16(
            false, qf[kc], false, bfm, (short)0, acc, false, false);
      }
      for (int r = 0; r < 8; ++r)
        atomicAdd(&sLog[(r + 8 * khalf) * 32 + t2 * 16 + row16], acc[r]);
    }
    __syncthreads();

    // Denominator (logits are cosines in [-1,1]; plain exp is safe).
    if (w == 0 && lane < 16) {
      float d = 0.f;
      for (int j = 0; j < 32; ++j) d += __expf(sLog[lane * 32 + j]);
      denom += d;
    }

    // Build P A-fragment (exp of logits) directly in registers.
    union { v16bf v; __bf16 e[16]; } pu;
    for (int jj = 0; jj < 8; ++jj) {
      pu.e[jj]     = (__bf16)__expf(sLog[row16 * 32 + khalf * 8 + jj]);
      pu.e[8 + jj] = (__bf16)__expf(sLog[row16 * 32 + 16 + khalf * 8 + jj]);
    }

    // Accumulate M += P*V and W2 += P*V^2 over this wave's channels.
    for (int t = 0; t < NT; ++t) {
      int ch = cb + t * 16 + row16;
      v16bf vf = load_frag(sVt, 32, ch, 0, khalf);
      union { v16bf v; __bf16 e[16]; } v2u;
      v2u.v = vf;
      for (int i2 = 0; i2 < 16; ++i2) {
        float x = (float)v2u.e[i2];
        v2u.e[i2] = (__bf16)(x * x);
      }
      Macc[t] = __builtin_amdgcn_wmma_f32_16x16x32_bf16(
          false, pu.v, false, vf, (short)0, Macc[t], false, false);
      Wacc[t] = __builtin_amdgcn_wmma_f32_16x16x32_bf16(
          false, pu.v, false, v2u.v, (short)0, Wacc[t], false, false);
    }
  }

  __syncthreads();
  if (w == 0 && lane < 16) sDen[lane] = denom;
  __syncthreads();

  // Finalize: S*nc + M, squared error, block-level partial loss.
  float lsum = 0.f;
  for (int t = 0; t < NT; ++t) {
    int ch = cb + t * 16 + row16;
    for (int r = 0; r < 8; ++r) {
      int qr = r + 8 * khalf;          // C/D layout: M = r (+8 for upper lanes)
      int n  = n0 + qr;
      size_t g = ((size_t)b * N + n) * C + ch;
      float den = sDen[qr];
      float Mv  = Macc[t][r] / den;
      float Wv  = Wacc[t][r] / den;
      float S2  = Wv - Mv * Mv;
      float S   = sqrtf(fmaxf(S2, 0.f) + 1e-12f);
      float nc  = (cf[g] - meanC[b * C + ch]) * rstdC[b * C + ch];
      float o   = S * nc + Mv;
      float d   = cs[g] - o;
      lsum += d * d;
    }
  }
  lsum *= scale;
  for (int off = 16; off; off >>= 1) lsum += __shfl_down(lsum, off, 32);
  if (lane == 0) atomicAdd(out, lsum);
}

extern "C" void kernel_launch(void* const* d_in, const int* in_sizes, int n_in,
                              void* d_out, int out_size, void* d_ws, size_t ws_size,
                              hipStream_t stream) {
  (void)in_sizes; (void)n_in; (void)out_size; (void)ws_size;
  const float* cs[3]  = {(const float*)d_in[0], (const float*)d_in[1], (const float*)d_in[2]};
  const float* cf[3]  = {(const float*)d_in[3], (const float*)d_in[4], (const float*)d_in[5]};
  const float* sfp[3] = {(const float*)d_in[6], (const float*)d_in[7], (const float*)d_in[8]};
  const int B = 4;
  const int Ns[3] = {64 * 64, 32 * 32, 16 * 16};
  const int Cs[3] = {256, 512, 512};

  zero_kernel<<<1, 32, 0, stream>>>((float*)d_out);

  float* w = (float*)d_ws;
  for (int l = 0; l < 3; ++l) {
    const int N = Ns[l], C = Cs[l];
    float* meanC = w; w += B * C;
    float* rstdC = w; w += B * C;
    float* meanS = w; w += B * C;
    float* rstdS = w; w += B * C;
    float* qinv  = w; w += B * N;
    float* kinv  = w; w += B * N;

    stats_kernel<<<B * C, 256, 0, stream>>>(cf[l],  meanC, rstdC, N, C);
    stats_kernel<<<B * C, 256, 0, stream>>>(sfp[l], meanS, rstdS, N, C);

    const int rows = B * N;
    rownorm_kernel<<<(rows + 7) / 8, 256, 0, stream>>>(cf[l],  meanC, rstdC, qinv, N, C, rows);
    rownorm_kernel<<<(rows + 7) / 8, 256, 0, stream>>>(sfp[l], meanS, rstdS, kinv, N, C, rows);

    const float scale = 1.0f / ((float)B * (float)N * (float)C);
    dim3 grid(B * (N / 16));
    if (C == 256) {
      attn_kernel<256><<<grid, 128, 0, stream>>>(cs[l], cf[l], sfp[l],
          meanC, rstdC, meanS, rstdS, qinv, kinv, (float*)d_out, N, scale);
    } else {
      attn_kernel<512><<<grid, 128, 0, stream>>>(cs[l], cf[l], sfp[l],
          meanC, rstdC, meanS, rstdS, qinv, kinv, (float*)d_out, N, scale);
    }
  }
}
